// DP3ObsEncoder_86406152061395
// MI455X (gfx1250) — matrix operand order repaired and checked by reading, not compiled, not run on hardware
//
#include <hip/hip_runtime.h>

typedef __attribute__((ext_vector_type(16))) _Float16 v16h;
typedef __attribute__((ext_vector_type(8)))  _Float16 v8h;
typedef __attribute__((ext_vector_type(8)))  float    v8f;

#define NPC 8192        // points per cloud (reference fixed)
#define THREADS 256
#define PPT (NPC / THREADS)   // 32 points per thread
#define EPSF 1e-5f

// ---------------------------------------------------------------------------
// Kernel 1: farthest point sampling. One block per batch element.
// Coordinates live in registers (LDS copy only for centroid gather).
// Argmax: wave32 shuffle reduce + 8-entry LDS combine; 2 barriers/step.
// ---------------------------------------------------------------------------
__global__ void __launch_bounds__(THREADS)
fps_kernel(const float* __restrict__ pc, const int* __restrict__ npts_p,
           float* __restrict__ samp) {
  extern __shared__ char smem[];
  float* px   = (float*)smem;            // 8192
  float* py   = px + NPC;                // 8192
  float* pz   = py + NPC;                // 8192
  float* rv   = pz + NPC;                // 8 (per-wave best)
  int*   ri   = (int*)(rv + 8);          // 8
  float* cent = (float*)(ri + 8);        // 4

  const int b    = blockIdx.x;
  const int tid  = threadIdx.x;
  const int lane = tid & 31;
  const int wave = tid >> 5;
  const int np   = npts_p[0];
  const float* pcb = pc + (size_t)b * NPC * 3;

  // register-resident coordinates; mirror into LDS for the random gather
  float lx[PPT], ly[PPT], lz[PPT], dist[PPT];
#pragma unroll
  for (int i = 0; i < PPT; ++i) {
    const int p = tid + (i << 8);
    const float x = pcb[p * 3 + 0];
    const float y = pcb[p * 3 + 1];
    const float z = pcb[p * 3 + 2];
    lx[i] = x; ly[i] = y; lz[i] = z;
    px[p] = x; py[p] = y; pz[p] = z;
    dist[i] = __builtin_inff();
  }
  __syncthreads();

  int far = 0;
  for (int t = 0; t < np; ++t) {
    if (tid == 0) {
      const float cx = px[far], cy = py[far], cz = pz[far];
      cent[0] = cx; cent[1] = cy; cent[2] = cz;
      float* o = samp + ((size_t)b * np + t) * 3;
      o[0] = cx; o[1] = cy; o[2] = cz;   // reference emits centroid of each step
    }
    __syncthreads();
    const float cx = cent[0], cy = cent[1], cz = cent[2];

    float best = -1.0f;
    int   bidx = 0;
#pragma unroll
    for (int i = 0; i < PPT; ++i) {
      const float dx = lx[i] - cx, dy = ly[i] - cy, dz = lz[i] - cz;
      const float d  = dx * dx + dy * dy + dz * dz;
      const float nd = fminf(dist[i], d);
      dist[i] = nd;
      if (nd > best) { best = nd; bidx = tid + (i << 8); }  // strict >: first max
    }
    // wave32 argmax reduce (tie-break: smaller index)
#pragma unroll
    for (int m = 16; m > 0; m >>= 1) {
      const float ov = __shfl_xor(best, m, 32);
      const int   oi = __shfl_xor(bidx, m, 32);
      if (ov > best || (ov == best && oi < bidx)) { best = ov; bidx = oi; }
    }
    if (lane == 0) { rv[wave] = best; ri[wave] = bidx; }
    __syncthreads();
    // every thread combines the 8 wave results identically
    float fb = rv[0]; int fi = ri[0];
#pragma unroll
    for (int w = 1; w < 8; ++w) {
      const float ov = rv[w]; const int oi = ri[w];
      if (ov > fb || (ov == fb && oi < fi)) { fb = ov; fi = oi; }
    }
    far = fi;   // next rv/ri writes happen only after the next barrier
  }
}

// ---------------------------------------------------------------------------
// WMMA fragment helpers — CDNA5 ISA 7.12.2 layouts (wave32), vectorized:
// every fragment is fetched with two 16-byte LDS loads (ds_load_b128).
// ---------------------------------------------------------------------------
union fragu { v16h v; v8h h[2]; };

// A 16x32 f16 from row-major activations h[16][K].
__device__ inline v16h load_a_frag(const _Float16* __restrict__ h, int K, int k0, int lane) {
  const int m  = lane & 15;
  const int hi = lane >> 4;
  const _Float16* base = h + m * K + k0 + hi * 8;
  fragu f;
  f.h[0] = *(const v8h*)(base);
  f.h[1] = *(const v8h*)(base + 16);
  return f.v;
}

// B 32x16 f16 from TRANSPOSED weights wT[N][K] (two 8-half runs per lane: k=8h.., 16+8h..).
__device__ inline v16h load_b_frag(const _Float16* __restrict__ wT, int K, int k0, int n0, int lane) {
  const int n  = n0 + (lane & 15);
  const _Float16* base = wT + n * K + k0 + (lane >> 4) * 8;
  fragu f;
  f.h[0] = *(const v8h*)(base);
  f.h[1] = *(const v8h*)(base + 16);
  return f.v;
}

// C/D 16x16 f32: lane L, reg r -> m = 8*(L>>4)+r, n = n0+(L&15).
__device__ inline void store_c_bias(float* __restrict__ pre, int Nn, int n0, int lane,
                                    v8f c, const float* __restrict__ bias) {
  const int n  = n0 + (lane & 15);
  const int m0 = (lane >> 4) * 8;
  const float bs = bias[n];
#pragma unroll
  for (int r = 0; r < 8; ++r) pre[(m0 + r) * Nn + n] = c[r] + bs;
}

// ---------------------------------------------------------------------------
// Wave-parallel LayerNorm+ReLU over one row of width W (lane-strided, shfl).
// ---------------------------------------------------------------------------
__device__ inline float ln_wave_stats(const float* __restrict__ row, int W, int lane,
                                      float* mean_out) {
  float s = 0.f;
  for (int i = lane; i < W; i += 32) s += row[i];
#pragma unroll
  for (int m = 16; m > 0; m >>= 1) s += __shfl_xor(s, m, 32);
  const float mean = s / (float)W;
  float v = 0.f;
  for (int i = lane; i < W; i += 32) { const float d = row[i] - mean; v += d * d; }
#pragma unroll
  for (int m = 16; m > 0; m >>= 1) v += __shfl_xor(v, m, 32);
  *mean_out = mean;
  return rsqrtf(v / (float)W + EPSF);
}

__device__ inline void ln_relu_wave_h(const float* __restrict__ row, int W,
                                      const float* __restrict__ g, const float* __restrict__ be,
                                      _Float16* __restrict__ outh, int lane) {
  float mean;
  const float rs = ln_wave_stats(row, W, lane, &mean);
  for (int i = lane; i < W; i += 32) {
    const float y = (row[i] - mean) * rs * g[i] + be[i];
    outh[i] = (_Float16)fmaxf(y, 0.f);
  }
}

__device__ inline void ln_relu_wave_f(float* __restrict__ row, int W,
                                      const float* __restrict__ g, const float* __restrict__ be,
                                      int lane) {
  float mean;
  const float rs = ln_wave_stats(row, W, lane, &mean);
  for (int i = lane; i < W; i += 32) {
    const float y = (row[i] - mean) * rs * g[i] + be[i];
    row[i] = fmaxf(y, 0.f);
  }
}

// ---------------------------------------------------------------------------
// Kernel 2: fused point MLP (WMMA f16->f32) + LN/ReLU + max-pool + heads.
// One block (8 waves) per batch element; weights staged TRANSPOSED (N-major)
// as f16 in LDS so B fragments are contiguous 128-bit LDS loads.
// ---------------------------------------------------------------------------
__global__ void __launch_bounds__(THREADS)
mlp_kernel(const float* __restrict__ samp, const int* __restrict__ npts_p,
           const float* __restrict__ joint,
           const float* __restrict__ w1, const float* __restrict__ b1,
           const float* __restrict__ g1, const float* __restrict__ be1,
           const float* __restrict__ w2, const float* __restrict__ b2,
           const float* __restrict__ g2, const float* __restrict__ be2,
           const float* __restrict__ w3, const float* __restrict__ b3,
           const float* __restrict__ g3, const float* __restrict__ be3,
           const float* __restrict__ wp, const float* __restrict__ bp,
           const float* __restrict__ gp, const float* __restrict__ bep,
           const float* __restrict__ ws1, const float* __restrict__ bs1,
           const float* __restrict__ ws2, const float* __restrict__ bs2,
           float* __restrict__ out) {
  extern __shared__ char smem[];
  _Float16* w1t = (_Float16*)smem;          // [64][32]   (K padded 3->32)
  _Float16* w2t = w1t + 64 * 32;            // [128][64]
  _Float16* w3t = w2t + 128 * 64;           // [256][128]
  _Float16* ah  = w3t + 256 * 128;          // [16][32]
  _Float16* h1h = ah + 16 * 32;             // [16][64]
  _Float16* h2h = h1h + 16 * 64;            // [16][128]
  float* pre  = (float*)(h2h + 16 * 128);   // [16][256] (reused per layer)
  float* gmax = pre + 16 * 256;             // 256
  float* cbuf = gmax + 256;                 // 1344: b1,g1,be1,b2,g2,be2,b3,g3,be3
  float* tvec = cbuf + 1344;                // 64
  float* svec = tvec + 64;                  // 64
  float* stat = svec + 64;                  // 2

  const int tid  = threadIdx.x;
  const int b    = blockIdx.x;
  const int lane = tid & 31;
  const int wave = tid >> 5;
  const int np   = npts_p[0];

  // ---- stage weights transposed (f32 -> f16) and constants into LDS ----
  for (int i = tid; i < 64 * 32; i += THREADS) {
    const int n = i & 63, k = i >> 6;
    w1t[n * 32 + k] = (k < 3) ? (_Float16)w1[k * 64 + n] : (_Float16)0.f;
  }
  for (int i = tid; i < 64 * 128; i += THREADS) {
    const int n = i & 127, k = i >> 7;
    w2t[n * 64 + k] = (_Float16)w2[k * 128 + n];
  }
  for (int i = tid; i < 128 * 256; i += THREADS) {
    const int n = i & 255, k = i >> 8;
    w3t[n * 128 + k] = (_Float16)w3[k * 256 + n];
  }
  if (tid < 64)  { cbuf[tid] = b1[tid]; cbuf[64 + tid] = g1[tid]; cbuf[128 + tid] = be1[tid]; }
  if (tid < 128) { cbuf[192 + tid] = b2[tid]; cbuf[320 + tid] = g2[tid]; cbuf[448 + tid] = be2[tid]; }
  { cbuf[576 + tid] = b3[tid]; cbuf[832 + tid] = g3[tid]; cbuf[1088 + tid] = be3[tid]; }
  gmax[tid] = 0.f;   // ReLU output => pooled max >= 0
  __syncthreads();

  const int ntiles = np >> 4;
  for (int tile = 0; tile < ntiles; ++tile) {
    const int p0 = tile << 4;

    // build A tile 16x32 f16 (only K=0..2 nonzero)
    for (int i = tid; i < 16 * 32; i += THREADS) {
      const int r = i >> 5, k = i & 31;
      ah[i] = (k < 3) ? (_Float16)samp[((size_t)b * np + p0 + r) * 3 + k] : (_Float16)0.f;
    }
    __syncthreads();

    // ---- layer 1: (16x32) x (32x64) ----
    if (wave < 4) {
      const int n0 = wave << 4;
      v16h a  = load_a_frag(ah, 32, 0, lane);
      v16h bb = load_b_frag(w1t, 32, 0, n0, lane);
      v8f  c  = {};
      c = __builtin_amdgcn_wmma_f32_16x16x32_f16(false, a, false, bb, (short)0, c, false, false);
      store_c_bias(pre, 64, n0, lane, c, cbuf + 0);
    }
    __syncthreads();
    for (int r = wave; r < 16; r += 8)
      ln_relu_wave_h(pre + r * 64, 64, cbuf + 64, cbuf + 128, h1h + r * 64, lane);
    __syncthreads();

    // ---- layer 2: (16x64) x (64x128) ----
    {
      const int n0 = wave << 4;
      v8f c = {};
#pragma unroll
      for (int kc = 0; kc < 2; ++kc) {
        v16h a  = load_a_frag(h1h, 64, kc * 32, lane);
        v16h bb = load_b_frag(w2t, 64, kc * 32, n0, lane);
        c = __builtin_amdgcn_wmma_f32_16x16x32_f16(false, a, false, bb, (short)0, c, false, false);
      }
      store_c_bias(pre, 128, n0, lane, c, cbuf + 192);
    }
    __syncthreads();
    for (int r = wave; r < 16; r += 8)
      ln_relu_wave_h(pre + r * 128, 128, cbuf + 320, cbuf + 448, h2h + r * 128, lane);
    __syncthreads();

    // ---- layer 3: (16x128) x (128x256), 2 N-tiles per wave ----
#pragma unroll
    for (int nt = 0; nt < 2; ++nt) {
      const int n0 = (wave + nt * 8) << 4;
      v8f c = {};
#pragma unroll
      for (int kc = 0; kc < 4; ++kc) {
        v16h a  = load_a_frag(h2h, 128, kc * 32, lane);
        v16h bb = load_b_frag(w3t, 128, kc * 32, n0, lane);
        c = __builtin_amdgcn_wmma_f32_16x16x32_f16(false, a, false, bb, (short)0, c, false, false);
      }
      store_c_bias(pre, 256, n0, lane, c, cbuf + 576);
    }
    __syncthreads();
    for (int r = wave; r < 16; r += 8)
      ln_relu_wave_f(pre + r * 256, 256, cbuf + 832, cbuf + 1088, lane);
    __syncthreads();

    // running max over this 16-point tile (thread per feature)
    {
      float m = gmax[tid];
#pragma unroll
      for (int p = 0; p < 16; ++p) m = fmaxf(m, pre[p * 256 + tid]);
      gmax[tid] = m;
    }
    __syncthreads();
  }

  // ---- global token head: LN(gmax @ wp + bp) ----
  if (tid < 64) {
    float acc = bp[tid];
    for (int i = 0; i < 256; ++i) acc += gmax[i] * wp[i * 64 + tid];
    tvec[tid] = acc;
  }
  __syncthreads();
  if (tid == 0) {
    float m = 0.f;
    for (int i = 0; i < 64; ++i) m += tvec[i];
    m /= 64.f;
    float v = 0.f;
    for (int i = 0; i < 64; ++i) { const float d = tvec[i] - m; v += d * d; }
    v /= 64.f;
    stat[0] = m;
    stat[1] = rsqrtf(v + EPSF);
  }
  __syncthreads();
  if (tid < 64)
    out[(size_t)b * 128 + tid] = (tvec[tid] - stat[0]) * stat[1] * gp[tid] + bep[tid];

  // ---- state MLP head: relu(js@ws1+bs1)@ws2+bs2 ----
  if (tid < 64) {
    float acc = bs1[tid];
    const float* jsb = joint + (size_t)b * 32;
    for (int i = 0; i < 32; ++i) acc += jsb[i] * ws1[i * 64 + tid];
    svec[tid] = fmaxf(acc, 0.f);
  }
  __syncthreads();
  if (tid < 64) {
    float acc = bs2[tid];
    for (int i = 0; i < 64; ++i) acc += svec[i] * ws2[i * 64 + tid];
    out[(size_t)b * 128 + 64 + tid] = acc;
  }
}

// ---------------------------------------------------------------------------
extern "C" void kernel_launch(void* const* d_in, const int* in_sizes, int n_in,
                              void* d_out, int out_size, void* d_ws, size_t ws_size,
                              hipStream_t stream) {
  (void)n_in; (void)out_size; (void)ws_size;
  const float* pc    = (const float*)d_in[0];
  const float* joint = (const float*)d_in[1];
  const int*   npts  = (const int*)d_in[2];
  const float *w1 = (const float*)d_in[3],  *b1 = (const float*)d_in[4];
  const float *g1 = (const float*)d_in[5],  *be1 = (const float*)d_in[6];
  const float *w2 = (const float*)d_in[7],  *b2 = (const float*)d_in[8];
  const float *g2 = (const float*)d_in[9],  *be2 = (const float*)d_in[10];
  const float *w3 = (const float*)d_in[11], *b3 = (const float*)d_in[12];
  const float *g3 = (const float*)d_in[13], *be3 = (const float*)d_in[14];
  const float *wp = (const float*)d_in[15], *bp = (const float*)d_in[16];
  const float *gp = (const float*)d_in[17], *bep = (const float*)d_in[18];
  const float *ws1 = (const float*)d_in[19], *bs1 = (const float*)d_in[20];
  const float *ws2 = (const float*)d_in[21], *bs2 = (const float*)d_in[22];
  float* out  = (float*)d_out;
  float* samp = (float*)d_ws;                 // B * num_points * 3 floats

  const int B = in_sizes[1] / 32;             // joint_state is (B, 32)

  const size_t fps_lds = (size_t)(3 * NPC) * 4 + 8 * 4 + 8 * 4 + 16;
  fps_kernel<<<B, THREADS, fps_lds, stream>>>(pc, npts, samp);

  const size_t mlp_lds =
      (size_t)2 * (64 * 32 + 128 * 64 + 256 * 128 + 16 * 32 + 16 * 64 + 16 * 128) +
      (size_t)4 * (16 * 256 + 256 + 1344 + 64 + 64 + 8);
  mlp_kernel<<<B, THREADS, mlp_lds, stream>>>(samp, npts, joint,
                                              w1, b1, g1, be1,
                                              w2, b2, g2, be2,
                                              w3, b3, g3, be3,
                                              wp, bp, gp, bep,
                                              ws1, bs1, ws2, bs2, out);
}
